// DotProductAttention_8847632629896
// MI455X (gfx1250) — matrix operand order repaired and checked by reading, not compiled, not run on hardware
//
#include <hip/hip_runtime.h>
#include <hip/hip_bf16.h>

// Problem constants (from reference)
#define BATCH 16
#define QLEN  2048
#define KLEN  2048
#define DDIM  128

typedef __attribute__((ext_vector_type(16))) __bf16 v16bf;
typedef __attribute__((ext_vector_type(8)))  __bf16 v8bf;
typedef __attribute__((ext_vector_type(8)))  float  v8f;

// Native fp32 -> bf16 (RNE); lets the backend pick the hw convert op.
static __device__ __forceinline__ __bf16 f2bf(float f) {
    return static_cast<__bf16>(f);
}

// LDS strides (elements). Chosen for 16B/32B-aligned fragment loads + bank spread.
#define SV_STRIDE 80   // V transposed: [feature 0..127][key 0..63 + pad]
#define SP_STRIDE 72   // per-wave P buffer: [row 0..15][key 0..63 + pad]

__global__ __launch_bounds__(256)
void flash_attn_bf16_wmma(const float* __restrict__ Qp,
                          const float* __restrict__ Kp,
                          const float* __restrict__ Vp,
                          const int*   __restrict__ VLp,
                          float* __restrict__ Op) {
    __shared__ __bf16 sK[64 * DDIM];          // 64 keys x 128 feats, row-major (16KB)
    __shared__ __bf16 sV[DDIM * SV_STRIDE];   // transposed V block (20KB)
    __shared__ __bf16 sP[8][16 * SP_STRIDE];  // per-wave P transpose buffer (18KB)

    const int tid  = threadIdx.x;
    const int wave = tid >> 5;
    const int lane = tid & 31;
    const int g    = lane >> 4;   // lane group (0: lanes 0-15, 1: lanes 16-31)
    const int ln   = lane & 15;

    const int b     = blockIdx.y;
    const int qbase = blockIdx.x * 128 + wave * 16;

    const float* Qrow = Qp + ((size_t)b * QLEN + qbase + ln) * DDIM;
    const float* Kg   = Kp + (size_t)b * KLEN * DDIM;
    const float* Vg   = Vp + (size_t)b * KLEN * DDIM;
    float*       Ob   = Op + ((size_t)b * QLEN + qbase) * DDIM;

    const int vl   = VLp[b];
    const int kend = ((vl + 63) & ~63) < KLEN ? ((vl + 63) & ~63) : KLEN;

    // Softmax in the log2 domain: scores arrive pre-multiplied by (1/sqrt(d))*log2(e)
    // because Q is scaled before bf16 conversion. Masked score = -1e6 * log2(e).
    const float qscale = 0.08838834764831845f * 1.4426950408889634f;
    const float MASKED = -1.4426950408889634e6f;

    // ---- Load Q tile as 4 bf16 A-fragments (16x32 each), pre-scaled ----
    // A fragment (16x32), contiguous k order used by this code: lane holds row M=ln; elems j -> K=g*16+j.
    // (c gives the 32-wide K chunk.)
    v16bf qa[4];
#pragma unroll
    for (int c = 0; c < 4; ++c) {
        const float4 r0 = *(const float4*)(Qrow + c * 32 + g * 16);
        const float4 r1 = *(const float4*)(Qrow + c * 32 + g * 16 + 4);
        const float4 r2 = *(const float4*)(Qrow + c * 32 + g * 16 + 8);
        const float4 r3 = *(const float4*)(Qrow + c * 32 + g * 16 + 12);
        qa[c][0]  = f2bf(r0.x * qscale); qa[c][1]  = f2bf(r0.y * qscale);
        qa[c][2]  = f2bf(r0.z * qscale); qa[c][3]  = f2bf(r0.w * qscale);
        qa[c][4]  = f2bf(r1.x * qscale); qa[c][5]  = f2bf(r1.y * qscale);
        qa[c][6]  = f2bf(r1.z * qscale); qa[c][7]  = f2bf(r1.w * qscale);
        qa[c][8]  = f2bf(r2.x * qscale); qa[c][9]  = f2bf(r2.y * qscale);
        qa[c][10] = f2bf(r2.z * qscale); qa[c][11] = f2bf(r2.w * qscale);
        qa[c][12] = f2bf(r3.x * qscale); qa[c][13] = f2bf(r3.y * qscale);
        qa[c][14] = f2bf(r3.z * qscale); qa[c][15] = f2bf(r3.w * qscale);
    }

    // ---- Online-softmax state (log2 domain for m) ----
    float mrow[8], lrow[8];
    v8f   o[8];
#pragma unroll
    for (int r = 0; r < 8; ++r) { mrow[r] = -3.0e38f; lrow[r] = 0.0f; }
#pragma unroll
    for (int ft = 0; ft < 8; ++ft) o[ft] = (v8f){};

    // Staging roles: each of 256 threads moves 32 contiguous floats of K and V.
    const int srow = tid >> 2;         // 0..63 key row
    const int sfb  = (tid & 3) * 32;   // feature base

    for (int kv = 0; kv < kend; kv += 64) {
        __syncthreads();  // previous iteration done reading sK/sV

        // ---- Cooperative staging: K row-major, V transposed, fp32->bf16 ----
        {
            const float4* ks = (const float4*)(Kg + (size_t)(kv + srow) * DDIM + sfb);
            const float4* vs = (const float4*)(Vg + (size_t)(kv + srow) * DDIM + sfb);
            __bf16* kd = &sK[srow * DDIM + sfb];
#pragma unroll
            for (int i = 0; i < 8; ++i) {
                float4 kvec = ks[i];
                kd[i * 4 + 0] = f2bf(kvec.x); kd[i * 4 + 1] = f2bf(kvec.y);
                kd[i * 4 + 2] = f2bf(kvec.z); kd[i * 4 + 3] = f2bf(kvec.w);
                float4 vvec = vs[i];
                sV[(sfb + i * 4 + 0) * SV_STRIDE + srow] = f2bf(vvec.x);
                sV[(sfb + i * 4 + 1) * SV_STRIDE + srow] = f2bf(vvec.y);
                sV[(sfb + i * 4 + 2) * SV_STRIDE + srow] = f2bf(vvec.z);
                sV[(sfb + i * 4 + 3) * SV_STRIDE + srow] = f2bf(vvec.w);
            }
            if (kv + 64 < kend) {  // gfx1250 global_prefetch_b8 for next block
                __builtin_prefetch(Kg + (size_t)(kv + 64 + srow) * DDIM + sfb, 0, 0);
                __builtin_prefetch(Vg + (size_t)(kv + 64 + srow) * DDIM + sfb, 0, 0);
            }
        }
        __syncthreads();

        // ---- S = Q @ K^T for four 16-key tiles; mask; track running max ----
        float sc[4][8];
        float mx[8];
#pragma unroll
        for (int r = 0; r < 8; ++r) mx[r] = -3.0e38f;
#pragma unroll
        for (int t = 0; t < 4; ++t) {
            v8f acc = (v8f){};
#pragma unroll
            for (int c = 0; c < 4; ++c) {
                // B layout: lane -> key column N = t*16+ln; elems j -> K = c*32+g*16+j
                v16bf kb = *(const v16bf*)&sK[(t * 16 + ln) * DDIM + c * 32 + g * 16];
                acc = __builtin_amdgcn_wmma_f32_16x16x32_bf16(
                    false, qa[c], false, kb, (short)0, acc, false, false);
            }
            const bool ok = (kv + t * 16 + ln) < vl;
#pragma unroll
            for (int r = 0; r < 8; ++r) {
                sc[t][r] = ok ? acc[r] : MASKED;
                mx[r]    = fmaxf(mx[r], sc[t][r]);
            }
        }

        // ---- row max over 16-lane half (C-layout: lanes 0-15 hold M=g*8+r) ----
#pragma unroll
        for (int off = 1; off < 16; off <<= 1)
#pragma unroll
            for (int r = 0; r < 8; ++r)
                mx[r] = fmaxf(mx[r], __shfl_xor(mx[r], off, 32));

        float alpha[8], psum[8];
#pragma unroll
        for (int r = 0; r < 8; ++r) {
            float mn = fmaxf(mrow[r], mx[r]);
            alpha[r] = exp2f(mrow[r] - mn);
            mrow[r]  = mn;
            psum[r]  = 0.0f;
        }

        // ---- p = exp2(s - m); write to per-wave LDS in C-layout ----
        __bf16* pb = sP[wave];
#pragma unroll
        for (int t = 0; t < 4; ++t)
#pragma unroll
            for (int r = 0; r < 8; ++r) {
                float p = exp2f(sc[t][r] - mrow[r]);
                psum[r] += p;
                pb[(g * 8 + r) * SP_STRIDE + t * 16 + ln] = f2bf(p);
            }
#pragma unroll
        for (int off = 1; off < 16; off <<= 1)
#pragma unroll
            for (int r = 0; r < 8; ++r)
                psum[r] += __shfl_xor(psum[r], off, 32);
#pragma unroll
        for (int r = 0; r < 8; ++r) lrow[r] = lrow[r] * alpha[r] + psum[r];
#pragma unroll
        for (int ft = 0; ft < 8; ++ft)
#pragma unroll
            for (int r = 0; r < 8; ++r) o[ft][r] *= alpha[r];

        // ---- read P back as two A-fragments (16x32 each) ----
        v16bf pa[2];
#pragma unroll
        for (int kc = 0; kc < 2; ++kc) {
            v8bf plo = *(const v8bf*)&pb[ln * SP_STRIDE + kc * 32 + g * 16];
            v8bf phi = *(const v8bf*)&pb[ln * SP_STRIDE + kc * 32 + g * 16 + 8];
            pa[kc] = __builtin_shufflevector(plo, phi,
                0, 1, 2, 3, 4, 5, 6, 7, 8, 9, 10, 11, 12, 13, 14, 15);
        }

        // ---- O += P @ V  (8 feature tiles x 2 key chunks) ----
#pragma unroll
        for (int ft = 0; ft < 8; ++ft)
#pragma unroll
            for (int kc = 0; kc < 2; ++kc) {
                // B layout: lane -> feature col N = ft*16+ln; elems j -> key kc*32+g*16+j
                v16bf vb = *(const v16bf*)&sV[(ft * 16 + ln) * SV_STRIDE + kc * 32 + g * 16];
                o[ft] = __builtin_amdgcn_wmma_f32_16x16x32_bf16(
                    false, pa[kc], false, vb, (short)0, o[ft], false, false);
            }
    }

    // ---- epilogue: normalize and store ----
    float inv[8];
#pragma unroll
    for (int r = 0; r < 8; ++r) inv[r] = 1.0f / lrow[r];
#pragma unroll
    for (int ft = 0; ft < 8; ++ft)
#pragma unroll
        for (int r = 0; r < 8; ++r)
            Ob[(size_t)(g * 8 + r) * DDIM + ft * 16 + ln] = o[ft][r] * inv[r];
}

extern "C" void kernel_launch(void* const* d_in, const int* in_sizes, int n_in,
                              void* d_out, int out_size, void* d_ws, size_t ws_size,
                              hipStream_t stream) {
    const float* Qp = (const float*)d_in[0];
    const float* Kp = (const float*)d_in[1];
    const float* Vp = (const float*)d_in[2];
    const int*   VL = (const int*)d_in[3];
    float*       Op = (float*)d_out;

    dim3 grid(QLEN / 128, BATCH);  // 16 x 16 blocks, 8 waves per block
    flash_attn_bf16_wmma<<<grid, 256, 0, stream>>>(Qp, Kp, Vp, VL, Op);
}
